// RIFLEXAttention_18734647345228
// MI455X (gfx1250) — matrix operand order repaired and checked by reading, not compiled, not run on hardware
//
#include <hip/hip_runtime.h>
#include <hip/hip_bf16.h>
#include <math.h>

// RIFLEX attention for MI455X (gfx1250): bf16 WMMA GEMMs + flash attention.
// B=2, S=2048, D=1024, H=16, HD=64.

#define Bv 2
#define Sv 2048
#define Dv 1024
#define Hv 16
#define HDv 64

typedef __attribute__((ext_vector_type(8)))  float  v8f;
typedef __attribute__((ext_vector_type(16))) __bf16 v16bf;
typedef __attribute__((ext_vector_type(8)))  __bf16 v8bf;

__device__ inline v8f vzero8() {
  v8f v = {0.f,0.f,0.f,0.f,0.f,0.f,0.f,0.f};
  return v;
}

__device__ inline v16bf join8(v8bf lo, v8bf hi) {
  return __builtin_shufflevector(lo, hi, 0,1,2,3,4,5,6,7,8,9,10,11,12,13,14,15);
}

// A fragment (16-bit A 16x32): caller passes lane-resolved base (row*ld + kh)
__device__ inline v16bf load_afrag(const __bf16* p) {
  v8bf lo = *(const v8bf*)(p);
  v8bf hi = *(const v8bf*)(p + 8);
  return join8(lo, hi);
}

// ---------------------------------------------------------------- fp32 -> bf16
__global__ __launch_bounds__(256) void cvt_f32_bf16(const float* __restrict__ src,
                                                    __bf16* __restrict__ dst, int n) {
  int i = blockIdx.x * 256 + threadIdx.x;
  if (i < n) dst[i] = (__bf16)src[i];
}

// ---------------------------------------------------------------- GEMM Y = A (M=4096,K=1024) x W^T (N=1024) + bias
// Wave tile: 32(M) x 64(N), k-step 32, register double-buffered.
// mode 0: write bf16 scattered to [B,H,S,HD]   (Q, K)
// mode 1: write bf16 scattered to [B,H,HD,S]   (V transposed)
// mode 2: write fp32 to [M,N] row-major        (final output)
__global__ __launch_bounds__(256)
void gemm_bf16(const __bf16* __restrict__ A, const __bf16* __restrict__ W,
               const float* __restrict__ bias, void* __restrict__ out, int mode) {
  const int Kd = 1024, Nd = 1024;
  int lane = threadIdx.x & 31;
  int wid  = blockIdx.x * 8 + (threadIdx.x >> 5);  // 2048 waves total
  int mt = wid >> 4;            // 0..127  (M/32)
  int nt = wid & 15;            // 0..15   (N/64)
  int mbase = mt * 32;
  int nbase = nt * 64;

  // A fragment addressing (16-bit A 16x32): lane&15 = row, contiguous K halves
  int akh  = (lane >> 4) * 16;
  const __bf16* Abase0 = A + (size_t)(mbase + (lane & 15)) * Kd + akh;
  const __bf16* Abase1 = Abase0 + (size_t)16 * Kd;

  // B fragment addressing (16-bit B 32x16): lane&15 = col, contiguous K halves
  int bcol = nbase + (lane & 15);
  int bkh  = (lane >> 4) * 16;
  const __bf16* Bbase = W + (size_t)bcol * Kd + bkh;

  v8f acc[2][4];
#pragma unroll
  for (int i = 0; i < 2; ++i)
#pragma unroll
    for (int j = 0; j < 4; ++j) acc[i][j] = vzero8();

  // prologue loads (k = 0)
  v16bf a_cur[2], b_cur[4], a_nxt[2], b_nxt[4];
  a_cur[0] = load_afrag(Abase0);
  a_cur[1] = load_afrag(Abase1);
#pragma unroll
  for (int j = 0; j < 4; ++j)
    b_cur[j] = *(const v16bf*)(Bbase + (size_t)(j * 16) * Kd);

#pragma unroll 2
  for (int k0 = 0; k0 < Kd; k0 += 32) {
    int kn = k0 + 32;
    if (kn < Kd) {
      __builtin_prefetch(Abase0 + kn + 32, 0, 1);   // global_prefetch_b8
      a_nxt[0] = load_afrag(Abase0 + kn);
      a_nxt[1] = load_afrag(Abase1 + kn);
#pragma unroll
      for (int j = 0; j < 4; ++j)
        b_nxt[j] = *(const v16bf*)(Bbase + (size_t)(j * 16) * Kd + kn);
    }
#pragma unroll
    for (int i = 0; i < 2; ++i)
#pragma unroll
      for (int j = 0; j < 4; ++j)
        acc[i][j] = __builtin_amdgcn_wmma_f32_16x16x32_bf16(
            false, a_cur[i], false, b_cur[j], (short)0, acc[i][j], false, false);
#pragma unroll
    for (int i = 0; i < 2; ++i) a_cur[i] = a_nxt[i];
#pragma unroll
    for (int j = 0; j < 4; ++j) b_cur[j] = b_nxt[j];
  }

  // Epilogue: C layout = lane&15 -> col, (lane>>4)*8 + vgpr -> row
  int cg = lane >> 4, ccol = lane & 15;
#pragma unroll
  for (int i = 0; i < 2; ++i) {
#pragma unroll
    for (int j = 0; j < 4; ++j) {
      int n = nbase + j * 16 + ccol;
      float bv = bias[n];
#pragma unroll
      for (int r = 0; r < 8; ++r) {
        int m = mbase + i * 16 + cg * 8 + r;
        float val = acc[i][j][r] + bv;
        if (mode == 2) {
          ((float*)out)[(size_t)m * Nd + n] = val;
        } else {
          int b = m >> 11, s = m & 2047;
          int h = n >> 6,  d = n & 63;
          size_t idx = (mode == 0)
              ? ((((size_t)b * Hv + h) * Sv + s) * HDv + d)
              : ((((size_t)b * Hv + h) * HDv + d) * Sv + s);
          ((__bf16*)out)[idx] = (__bf16)val;
        }
      }
    }
  }
}

// ---------------------------------------------------------------- RoPE on Q and K in place ([B,H,S,HD] bf16)
__global__ __launch_bounds__(256)
void rope_kernel(__bf16* __restrict__ Q, __bf16* __restrict__ K) {
  int tid = blockIdx.x * 256 + threadIdx.x;   // B*H*S*32 = 2097152 threads
  int i  = tid & 31;
  int s  = (tid >> 5) & 2047;
  int bh = tid >> 16;                          // 0..31
  size_t base = ((size_t)bh * Sv + s) * HDv;
  float inv = __powf(10000.0f, -(float)i * (1.0f / 32.0f));
  float ang = (float)s * inv;
  float c = __cosf(ang), sn = __sinf(ang);
  float q0 = (float)Q[base + i], q1 = (float)Q[base + i + 32];
  Q[base + i]      = (__bf16)(q0 * c - q1 * sn);
  Q[base + i + 32] = (__bf16)(q1 * c + q0 * sn);
  float k0 = (float)K[base + i], k1 = (float)K[base + i + 32];
  K[base + i]      = (__bf16)(k0 * c - k1 * sn);
  K[base + i + 32] = (__bf16)(k1 * c + k0 * sn);
}

// ---------------------------------------------------------------- flash attention
// Q,K: [B,H,S,HD] bf16 ; Vt: [B,H,HD,S] bf16 ; mask: [B,1,S,S] f32 ; Ctx: [B,S,D] bf16
__global__ __launch_bounds__(256)
void flash_attn(const __bf16* __restrict__ Q, const __bf16* __restrict__ K,
                const __bf16* __restrict__ Vt, const float* __restrict__ mask,
                __bf16* __restrict__ Ctx) {
  __shared__ __bf16 pbuf[8][16 * 32];         // per-wave P staging (C-frag -> A-frag relayout)
  const float scale = 0.0163942637f;          // HD^-0.5 / ln(S) = 0.125 / ln(2048)

  int lane = threadIdx.x & 31;
  int w    = threadIdx.x >> 5;
  int wid  = blockIdx.x * 8 + w;              // 4096 waves
  int qt = wid & 127;                         // S/16 q-tiles
  int bh = wid >> 7;                          // 0..31
  int b = bh >> 4, h = bh & 15;
  int qbase = qt * 16;

  const __bf16* Qp = Q  + ((size_t)bh * Sv + qbase) * HDv;
  const __bf16* Kp = K  + (size_t)bh * Sv * HDv;
  const __bf16* Vp = Vt + (size_t)bh * HDv * Sv;
  const float*  Mp = mask + (size_t)b * Sv * Sv + (size_t)qbase * Sv;

  int arow = lane & 15;
  int akh  = (lane >> 4) * 16;
  int bcol = lane & 15;
  int bkh  = (lane >> 4) * 16;
  int cg = lane >> 4, ccol = lane & 15;

  // Q A-fragments for the two K-dim steps (HD=64 -> 2 x 32)
  v16bf qa[2];
#pragma unroll
  for (int t = 0; t < 2; ++t)
    qa[t] = load_afrag(Qp + (size_t)arow * HDv + t * 32 + akh);

  float mrow[8], lrow[8];
  v8f o[4];
#pragma unroll
  for (int r = 0; r < 8; ++r) { mrow[r] = -3.0e38f; lrow[r] = 0.f; }
#pragma unroll
  for (int j = 0; j < 4; ++j) o[j] = vzero8();

  for (int kb = 0; kb < Sv; kb += 32) {
    // ---- mask loads first (independent of score WMMAs -> overlap)
    float m0[8], m1[8];
#pragma unroll
    for (int r = 0; r < 8; ++r) {
      int qrow = cg * 8 + r;
      m0[r] = Mp[(size_t)qrow * Sv + kb + ccol];
      m1[r] = Mp[(size_t)qrow * Sv + kb + 16 + ccol];
    }

    // ---- S = Q . K^T for 32 keys (two 16-key score tiles)
    v8f s0 = vzero8(), s1 = vzero8();
#pragma unroll
    for (int t = 0; t < 2; ++t) {
      v16bf kf0 = *(const v16bf*)(Kp + (size_t)(kb + bcol) * HDv + t * 32 + bkh);
      v16bf kf1 = *(const v16bf*)(Kp + (size_t)(kb + 16 + bcol) * HDv + t * 32 + bkh);
      s0 = __builtin_amdgcn_wmma_f32_16x16x32_bf16(false, qa[t], false, kf0, (short)0, s0, false, false);
      s1 = __builtin_amdgcn_wmma_f32_16x16x32_bf16(false, qa[t], false, kf1, (short)0, s1, false, false);
    }

    // ---- V B-fragments early (independent of softmax -> overlap with VALU)
    v16bf vf[4];
#pragma unroll
    for (int j = 0; j < 4; ++j)
      vf[j] = *(const v16bf*)(Vp + (size_t)(j * 16 + bcol) * Sv + kb + bkh);

    // ---- online softmax (row stats per (group, vgpr); reduce across 16 lanes)
    float p0[8], p1[8], corr[8];
#pragma unroll
    for (int r = 0; r < 8; ++r) {
      float a0 = s0[r] * scale + m0[r];
      float a1 = s1[r] * scale + m1[r];
      float mx = fmaxf(a0, a1);
      mx = fmaxf(mx, __shfl_xor(mx, 1, 32));
      mx = fmaxf(mx, __shfl_xor(mx, 2, 32));
      mx = fmaxf(mx, __shfl_xor(mx, 4, 32));
      mx = fmaxf(mx, __shfl_xor(mx, 8, 32));
      float nm = fmaxf(mrow[r], mx);
      corr[r] = __expf(mrow[r] - nm);
      mrow[r] = nm;
      p0[r] = __expf(a0 - nm);
      p1[r] = __expf(a1 - nm);
      float sum = p0[r] + p1[r];
      sum += __shfl_xor(sum, 1, 32);
      sum += __shfl_xor(sum, 2, 32);
      sum += __shfl_xor(sum, 4, 32);
      sum += __shfl_xor(sum, 8, 32);
      lrow[r] = lrow[r] * corr[r] + sum;
    }
#pragma unroll
    for (int j = 0; j < 4; ++j)
#pragma unroll
      for (int r = 0; r < 8; ++r) o[j][r] *= corr[r];

    // ---- stage P (16x32) to LDS in row-major, re-read as A-fragment
    __bf16* pb = &pbuf[w][0];
#pragma unroll
    for (int r = 0; r < 8; ++r) {
      int m = cg * 8 + r;
      pb[m * 32 + ccol]      = (__bf16)p0[r];
      pb[m * 32 + 16 + ccol] = (__bf16)p1[r];
    }
    asm volatile("s_wait_dscnt 0" ::: "memory");
    v8bf plo = *(const v8bf*)(pb + arow * 32 + akh);
    v8bf phi = *(const v8bf*)(pb + arow * 32 + akh + 8);
    v16bf pf = join8(plo, phi);

    // ---- O += P . V  (V transposed: contiguous K in rows of Vt)
#pragma unroll
    for (int j = 0; j < 4; ++j)
      o[j] = __builtin_amdgcn_wmma_f32_16x16x32_bf16(false, pf, false, vf[j], (short)0, o[j], false, false);
  }

  // ---- epilogue: normalize and scatter to Ctx [B,S,D]
#pragma unroll
  for (int j = 0; j < 4; ++j) {
#pragma unroll
    for (int r = 0; r < 8; ++r) {
      int qrow = qbase + cg * 8 + r;
      int d = j * 16 + ccol;
      float val = o[j][r] / lrow[r];
      Ctx[((size_t)b * Sv + qrow) * Dv + h * HDv + d] = (__bf16)val;
    }
  }
}

// ---------------------------------------------------------------- launch
extern "C" void kernel_launch(void* const* d_in, const int* in_sizes, int n_in,
                              void* d_out, int out_size, void* d_ws, size_t ws_size,
                              hipStream_t stream) {
  const float* X    = (const float*)d_in[0];
  const float* mask = (const float*)d_in[1];
  const float* Wq   = (const float*)d_in[2];
  const float* bq   = (const float*)d_in[3];
  const float* Wk   = (const float*)d_in[4];
  const float* bk   = (const float*)d_in[5];
  const float* Wv   = (const float*)d_in[6];
  const float* bv   = (const float*)d_in[7];
  const float* Wo   = (const float*)d_in[8];
  const float* bo   = (const float*)d_in[9];

  char* ws = (char*)d_ws;
  const size_t MB = 1u << 20;
  __bf16* Xb  = (__bf16*)(ws);             // 8 MB  [4096,1024]
  __bf16* Wqb = (__bf16*)(ws + 8  * MB);   // 2 MB
  __bf16* Wkb = (__bf16*)(ws + 10 * MB);   // 2 MB
  __bf16* Wvb = (__bf16*)(ws + 12 * MB);   // 2 MB
  __bf16* Wob = (__bf16*)(ws + 14 * MB);   // 2 MB
  __bf16* Qb  = (__bf16*)(ws + 16 * MB);   // 8 MB  [B,H,S,HD]
  __bf16* Kb  = (__bf16*)(ws + 24 * MB);   // 8 MB  [B,H,S,HD]
  __bf16* Vt  = (__bf16*)(ws + 32 * MB);   // 8 MB  [B,H,HD,S]
  __bf16* Ctx = (__bf16*)(ws + 40 * MB);   // 8 MB  [B,S,D]

  const int NX = Bv * Sv * Dv;             // 4194304
  const int NW = Dv * Dv;                  // 1048576
  cvt_f32_bf16<<<(NX + 255) / 256, 256, 0, stream>>>(X,  Xb,  NX);
  cvt_f32_bf16<<<(NW + 255) / 256, 256, 0, stream>>>(Wq, Wqb, NW);
  cvt_f32_bf16<<<(NW + 255) / 256, 256, 0, stream>>>(Wk, Wkb, NW);
  cvt_f32_bf16<<<(NW + 255) / 256, 256, 0, stream>>>(Wv, Wvb, NW);
  cvt_f32_bf16<<<(NW + 255) / 256, 256, 0, stream>>>(Wo, Wob, NW);

  gemm_bf16<<<256, 256, 0, stream>>>(Xb, Wqb, bq, (void*)Qb, 0);
  gemm_bf16<<<256, 256, 0, stream>>>(Xb, Wkb, bk, (void*)Kb, 0);
  gemm_bf16<<<256, 256, 0, stream>>>(Xb, Wvb, bv, (void*)Vt, 1);

  rope_kernel<<<8192, 256, 0, stream>>>(Qb, Kb);

  flash_attn<<<512, 256, 0, stream>>>(Qb, Kb, Vt, mask, Ctx);

  gemm_bf16<<<256, 256, 0, stream>>>(Ctx, Wob, bo, d_out, 2);
}
